// Attention_90211493085692
// MI455X (gfx1250) — compile-verified, matrix operand order checked by reading, not run on hardware
//
#include <hip/hip_runtime.h>

typedef __attribute__((ext_vector_type(16))) _Float16 v16h;
typedef __attribute__((ext_vector_type(8)))  _Float16 v8h;
typedef __attribute__((ext_vector_type(8)))  float    v8f;

#define HN 4
#define CC 32
#define CQD 128
#define NND 256

// ---- operand loaders (gfx1250 16-bit A/B layout) ----------------------------
// A (16x32 MxK): lane L holds row m=L%16; element e holds K = ((e&8)<<1)+(e&7)+8*(L>>4).
// Per lane: 8 contiguous halfs at k0=8*(L>>4), and 8 contiguous at k0+16.
// B (32x16 KxN) mirrors A with N in place of M, so an [N][K] row-major source
// loads with the exact same code.
__device__ __forceinline__ v16h load_op_f16(const _Float16* __restrict__ p, int ld) {
  const int lane = threadIdx.x & 31;
  const int m  = lane & 15;
  const int h8 = (lane >> 4) << 3;
  const _Float16* r = p + (size_t)m * ld + h8;
  union { v16h v; v8h h[2]; } u;
  u.h[0] = *reinterpret_cast<const v8h*>(r);
  u.h[1] = *reinterpret_cast<const v8h*>(r + 16);
  return u.v;
}

__device__ __forceinline__ v16h load_op_f32cvt(const float* __restrict__ p, int ld) {
  const int lane = threadIdx.x & 31;
  const int m  = lane & 15;
  const int h8 = (lane >> 4) << 3;
  const float* r = p + (size_t)m * ld + h8;
  v16h a;
  #pragma unroll
  for (int i = 0; i < 8; ++i) a[i]     = (_Float16)r[i];
  #pragma unroll
  for (int i = 0; i < 8; ++i) a[8 + i] = (_Float16)r[16 + i];
  return a;
}

__device__ __forceinline__ v8f wmma16(v16h a, v16h b, v8f c) {
  return __builtin_amdgcn_wmma_f32_16x16x32_f16(false, a, false, b, (short)0, c, false, false);
}

// ---- kernel 1: q/k/v/g projections, compile-time specialized per weight ----
// WI: 0=q (scaled), 1=k, 2=v (stored transposed), 3=g (sigmoid)
template<int WI>
__device__ __forceinline__ void proj_pass(const float* __restrict__ X,
                                          const float* __restrict__ W,
                                          _Float16* __restrict__ dst,
                                          _Float16* wlds)
{
  const int tid  = threadIdx.x;
  const int wave = tid >> 5;
  const int lane = tid & 31;
  const int half = lane >> 4;
  const int nlo  = lane & 15;
  const int M0   = blockIdx.x * 128 + wave * 16;

  for (int i = tid; i < CQD * CQD; i += 256) wlds[i] = (_Float16)W[i];
  __syncthreads();

  v16h aop[4];
  #pragma unroll
  for (int ks = 0; ks < 4; ++ks)
    aop[ks] = load_op_f32cvt(X + (size_t)M0 * CQD + ks * 32, CQD);

  #pragma unroll
  for (int jc = 0; jc < 8; ++jc) {
    v8f acc = {};
    #pragma unroll
    for (int ks = 0; ks < 4; ++ks) {
      v16h b = load_op_f16(wlds + (size_t)(jc * 16) * CQD + ks * 32, CQD);
      acc = wmma16(aop[ks], b, acc);
    }
    const int d  = jc * 16 + nlo;
    const int hh = d >> 5;
    const int c  = d & 31;
    #pragma unroll
    for (int r = 0; r < 8; ++r) {
      const int Mrow = M0 + r + 8 * half;
      const int n    = Mrow >> 8;
      const int row  = Mrow & 255;
      const float val = acc[r];
      if (WI == 0)
        dst[(((size_t)n * HN + hh) * NND + row) * CC + c] =
            (_Float16)(val * 0.17677669529663687f);              // q / sqrt(C)
      else if (WI == 1)
        dst[(((size_t)n * HN + hh) * NND + row) * CC + c] = (_Float16)val;
      else if (WI == 2)
        dst[(((size_t)n * HN + hh) * CC + c) * NND + row] = (_Float16)val; // v^T
      else
        dst[(size_t)Mrow * CQD + d] =
            (_Float16)(1.0f / (1.0f + __expf(-val)));            // sigmoid gate
    }
  }
  __syncthreads();
}

__global__ void __launch_bounds__(256)
proj_kernel(const float* __restrict__ qx, const float* __restrict__ kvx,
            const float* __restrict__ wq, const float* __restrict__ wk,
            const float* __restrict__ wv, const float* __restrict__ wg,
            _Float16* __restrict__ q_ws, _Float16* __restrict__ k_ws,
            _Float16* __restrict__ v_ws, _Float16* __restrict__ g_ws)
{
  __shared__ _Float16 wlds[CQD * CQD];          // 32 KB: one weight in f16
  proj_pass<0>(qx,  wq, q_ws, wlds);
  proj_pass<1>(kvx, wk, k_ws, wlds);
  proj_pass<2>(kvx, wv, v_ws, wlds);
  proj_pass<3>(qx,  wg, g_ws, wlds);
}

// ---- kernel 2: per-(n,h) attention with fused softmax + gating -------------
__global__ void __launch_bounds__(128)
attn_kernel(const _Float16* __restrict__ q_ws, const _Float16* __restrict__ k_ws,
            const _Float16* __restrict__ v_ws, const _Float16* __restrict__ g_ws,
            const float* __restrict__ mask_bias, const float* __restrict__ pair_bias,
            _Float16* __restrict__ o_ws)
{
  __shared__ _Float16 Plds[4][16 * 264];        // per-wave 16x256 probs (padded)
  const int tid  = threadIdx.x;
  const int wave = tid >> 5;
  const int lane = tid & 31;
  const int half = lane >> 4;
  const int nlo  = lane & 15;

  const int bx     = blockIdx.x;
  const int qchunk = bx & 3;
  const int h      = (bx >> 2) & 3;
  const int n      = bx >> 4;
  const int q0     = qchunk * 64 + wave * 16;

  const size_t base_qk = ((size_t)n * HN + h) * NND * CC;
  const size_t base_v  = ((size_t)n * HN + h) * CC * NND;

  // prefetch the V panel we will need after softmax (global_prefetch_b8)
  __builtin_prefetch(v_ws + base_v, 0, 0);
  __builtin_prefetch(v_ws + base_v + (size_t)16 * NND, 0, 0);

  v16h aq = load_op_f16(q_ws + base_qk + (size_t)q0 * CC, CC);

  // S = q k^T : K=32 -> one WMMA per 16x16 tile, 16 column tiles.
  // Bias is folded in right after each tile.
  v8f s[16];
  #pragma unroll
  for (int jc = 0; jc < 16; ++jc) {
    v16h bk = load_op_f16(k_ws + base_qk + (size_t)(jc * 16) * CC, CC);
    v8f z = {};
    s[jc] = wmma16(aq, bk, z);
    const int key = jc * 16 + nlo;
    const float mb = mask_bias[(size_t)n * NND + key];
    #pragma unroll
    for (int r = 0; r < 8; ++r) {
      const int qrow = q0 + r + 8 * half;
      s[jc][r] += mb + pair_bias[((size_t)h * NND + qrow) * NND + key];
    }
  }

  // softmax per row: a row lives in one VGPR across 16 lanes of a half
  float inv[8];
  #pragma unroll
  for (int r = 0; r < 8; ++r) {
    float mx = -3.0e38f;
    #pragma unroll
    for (int jc = 0; jc < 16; ++jc) mx = fmaxf(mx, s[jc][r]);
    #pragma unroll
    for (int mk = 1; mk < 16; mk <<= 1) mx = fmaxf(mx, __shfl_xor(mx, mk, 32));
    float sum = 0.0f;
    #pragma unroll
    for (int jc = 0; jc < 16; ++jc) { float e = __expf(s[jc][r] - mx); s[jc][r] = e; sum += e; }
    #pragma unroll
    for (int mk = 1; mk < 16; mk <<= 1) sum += __shfl_xor(sum, mk, 32);
    inv[r] = 1.0f / sum;
  }

  // P -> LDS (f16, row-major, ld=264) for re-load in A-operand layout
  _Float16* P = &Plds[wave][0];
  #pragma unroll
  for (int jc = 0; jc < 16; ++jc)
    #pragma unroll
    for (int r = 0; r < 8; ++r)
      P[(size_t)(r + 8 * half) * 264 + jc * 16 + nlo] = (_Float16)(s[jc][r] * inv[r]);

  __syncthreads();

  // O = P V : K=256 -> 8 k-steps, 2 column tiles (C=32)
  v8f o0 = {}, o1 = {};
  #pragma unroll
  for (int kt = 0; kt < 8; ++kt) {
    v16h ap  = load_op_f16(P + kt * 32, 264);
    v16h bv0 = load_op_f16(v_ws + base_v + kt * 32, NND);
    v16h bv1 = load_op_f16(v_ws + base_v + (size_t)16 * NND + kt * 32, NND);
    o0 = wmma16(ap, bv0, o0);
    o1 = wmma16(ap, bv1, o1);
  }

  // gate with sigmoid(g) and store merged-head f16
  #pragma unroll
  for (int r = 0; r < 8; ++r) {
    const int qrow = q0 + r + 8 * half;
    const size_t Mrow = (size_t)n * NND + qrow;
    {
      const int d = h * CC + nlo;
      const float g = (float)g_ws[Mrow * CQD + d];
      o_ws[Mrow * CQD + d] = (_Float16)(o0[r] * g);
    }
    {
      const int d = h * CC + 16 + nlo;
      const float g = (float)g_ws[Mrow * CQD + d];
      o_ws[Mrow * CQD + d] = (_Float16)(o1[r] * g);
    }
  }
}

// ---- kernel 3: output projection (O⊙g) @ w_o^T, f32 out --------------------
__global__ void __launch_bounds__(256)
outproj_kernel(const _Float16* __restrict__ o_ws, const float* __restrict__ wo,
               float* __restrict__ out)
{
  __shared__ _Float16 wlds[CQD * CQD];
  const int tid  = threadIdx.x;
  const int wave = tid >> 5;
  const int lane = tid & 31;
  const int half = lane >> 4;
  const int nlo  = lane & 15;
  const int M0   = blockIdx.x * 128 + wave * 16;

  for (int i = tid; i < CQD * CQD; i += 256) wlds[i] = (_Float16)wo[i];
  __syncthreads();

  v16h aop[4];
  #pragma unroll
  for (int ks = 0; ks < 4; ++ks)
    aop[ks] = load_op_f16(o_ws + (size_t)M0 * CQD + ks * 32, CQD);

  #pragma unroll
  for (int jc = 0; jc < 8; ++jc) {
    v8f acc = {};
    #pragma unroll
    for (int ks = 0; ks < 4; ++ks) {
      v16h b = load_op_f16(wlds + (size_t)(jc * 16) * CQD + ks * 32, CQD);
      acc = wmma16(aop[ks], b, acc);
    }
    #pragma unroll
    for (int r = 0; r < 8; ++r) {
      const int Mrow = M0 + r + 8 * half;
      out[(size_t)Mrow * CQD + jc * 16 + nlo] = acc[r];
    }
  }
}

// ---- launcher ---------------------------------------------------------------
extern "C" void kernel_launch(void* const* d_in, const int* in_sizes, int n_in,
                              void* d_out, int out_size, void* d_ws, size_t ws_size,
                              hipStream_t stream) {
  const float* qx        = (const float*)d_in[0];
  const float* kvx       = (const float*)d_in[1];
  const float* mask_bias = (const float*)d_in[2];
  const float* pair_bias = (const float*)d_in[3];
  const float* wq        = (const float*)d_in[4];
  const float* wk        = (const float*)d_in[5];
  const float* wv        = (const float*)d_in[6];
  const float* wg        = (const float*)d_in[7];
  const float* wo        = (const float*)d_in[8];
  float* out = (float*)d_out;

  const size_t M = (size_t)NND * NND;           // 65536 flattened (n, q) rows
  _Float16* q_ws = (_Float16*)d_ws;             // [n][h][q][c]   16 MB
  _Float16* k_ws = q_ws + M * CQD;              // [n][h][k][c]   16 MB
  _Float16* v_ws = k_ws + M * CQD;              // [n][h][c][k]   16 MB (transposed)
  _Float16* g_ws = v_ws + M * CQD;              // [M][128]       16 MB
  _Float16* o_ws = g_ws + M * CQD;              // [M][128]       16 MB

  proj_kernel<<<dim3(512), dim3(256), 0, stream>>>(qx, kvx, wq, wk, wv, wg,
                                                   q_ws, k_ws, v_ws, g_ws);
  attn_kernel<<<dim3(NND * HN * 4), dim3(128), 0, stream>>>(q_ws, k_ws, v_ws, g_ws,
                                                            mask_bias, pair_bias, o_ws);
  outproj_kernel<<<dim3(512), dim3(256), 0, stream>>>(o_ws, wo, out);
}
